// MultiHeadAttention_52879637348478
// MI455X (gfx1250) — matrix operand order repaired and checked by reading, not compiled, not run on hardware
//
#include <hip/hip_runtime.h>

// ---------------------------------------------------------------------------
// MultiHeadAttention forward for MI455X (gfx1250, wave32, WMMA).
// B=2, L=2048, D=1024, H=16, d_k=64. bf16 WMMA (16x16x32) with f32 accum.
// Pipeline:
//   0) f32->bf16 pre-convert of weights (Wq|Wk|Wv|Wo) and activations (q|k|v)
//   1) fused Q/K/V projection GEMM: async-LDS staged weight panels + WMMA
//   2) flash attention (online softmax), WMMA for QK^T and P*V
//   3) output projection GEMM -> f32 d_out
// ---------------------------------------------------------------------------

#define BB   2
#define LL   2048
#define DD   1024
#define HH   16
#define DK   64

typedef __attribute__((ext_vector_type(16))) __bf16       v16bf;
typedef __attribute__((ext_vector_type(8)))  float        v8f;
typedef __attribute__((ext_vector_type(8)))  unsigned int v8u;
typedef int v4i_async __attribute__((vector_size(16)));   // builtin's pointee

#define WMMA_BF16(A, B, C) \
  __builtin_amdgcn_wmma_f32_16x16x32_bf16(false, (A), false, (B), (short)0, (C), false, false)

#if defined(__has_builtin)
#if __has_builtin(__builtin_amdgcn_global_load_async_to_lds_b128) && \
    __has_builtin(__builtin_amdgcn_s_wait_asynccnt)
#define HAVE_ASYNC_LDS 1
#endif
#endif

__device__ __forceinline__ void copy_b128_to_lds(const unsigned int* g,
                                                 unsigned int* l) {
#ifdef HAVE_ASYNC_LDS
  // ASYNCcnt-tracked DMA: global -> LDS, no VGPR data path.
  __builtin_amdgcn_global_load_async_to_lds_b128((v4i_async*)g, (v4i_async*)l,
                                                 0, 0);
#else
  *reinterpret_cast<uint4*>(l) = *reinterpret_cast<const uint4*>(g);
#endif
}

#ifdef HAVE_ASYNC_LDS
#define WAIT_ASYNC(n) __builtin_amdgcn_s_wait_asynccnt(n)
#else
#define WAIT_ASYNC(n) ((void)0)
#endif

__device__ __forceinline__ unsigned int pack2_bf16(float x, float y) {
  unsigned xb = __builtin_bit_cast(unsigned, x);
  unsigned yb = __builtin_bit_cast(unsigned, y);
  xb = (xb + 0x7FFFu + ((xb >> 16) & 1u)) >> 16;          // RNE
  yb = (yb + 0x7FFFu + ((yb >> 16) & 1u)) & 0xFFFF0000u;  // RNE, keep hi half
  return (xb & 0xFFFFu) | yb;
}

__device__ __forceinline__ unsigned short cvt1_bf16(float x) {
  unsigned xb = __builtin_bit_cast(unsigned, x);
  xb = (xb + 0x7FFFu + ((xb >> 16) & 1u)) >> 16;
  return (unsigned short)xb;
}

__device__ __forceinline__ v16bf as_bf(v8u u) { return __builtin_bit_cast(v16bf, u); }

__device__ __forceinline__ v8f v8f_zero() {
  v8f z;
#pragma unroll
  for (int i = 0; i < 8; ++i) z[i] = 0.0f;
  return z;
}

// ---------------------------------------------------------------------------
// Kernel 0a: weights f32 -> bf16.  dst = Wq|Wk|Wv|Wo, 512K dwords each.
// ---------------------------------------------------------------------------
__global__ __launch_bounds__(256) void cvt_w_kernel(
    const float* __restrict__ Wq, const float* __restrict__ Wk,
    const float* __restrict__ Wv, const float* __restrict__ Wo,
    unsigned int* __restrict__ dst) {
  const int idx = blockIdx.x * blockDim.x + threadIdx.x;  // one dword (2 bf16)
  const int mat = idx >> 19;
  const int off = idx & ((1 << 19) - 1);
  const float* src = (mat == 0) ? Wq : (mat == 1) ? Wk : (mat == 2) ? Wv : Wo;
  float2 w2 = *reinterpret_cast<const float2*>(src + (size_t)off * 2);
  dst[idx] = pack2_bf16(w2.x, w2.y);
}

// ---------------------------------------------------------------------------
// Kernel 0b: activations f32 -> bf16.  dst = q|k|v, 2M dwords each.
// ---------------------------------------------------------------------------
__global__ __launch_bounds__(256) void cvt_x_kernel(
    const float* __restrict__ q_in, const float* __restrict__ k_in,
    const float* __restrict__ v_in, unsigned int* __restrict__ dst) {
  const int idx = blockIdx.x * blockDim.x + threadIdx.x;  // one dword (2 bf16)
  const int mat = idx >> 21;                              // 2M dwords / matrix
  const int off = idx & ((1 << 21) - 1);
  const float* src = (mat == 0) ? q_in : (mat == 1) ? k_in : v_in;
  float2 x2 = *reinterpret_cast<const float2*>(src + (size_t)off * 2);
  dst[idx] = pack2_bf16(x2.x, x2.y);
}

// ---------------------------------------------------------------------------
// Kernel 1: fused Q/K/V projections, Y = X @ W^T + b (all-bf16 operands,
// f32 accum).  Block = 8 waves sharing one 64-col weight panel; the panel is
// double-buffer staged into LDS with async global->LDS copies.
// Wave tile: 16 rows x 64 cols.  Q,K -> [B,H,L,dk] bf16;  V -> [B,H,dk,L].
// ---------------------------------------------------------------------------
#define WB_STRIDE 20   // dwords per staged row (16 data + 4 pad, 16B-aligned)

__global__ __launch_bounds__(256) void qkv_proj_kernel(
    const unsigned int* __restrict__ Xbf,   // q|k|v bf16, 2M dwords each
    const float* __restrict__ bq, const float* __restrict__ bk,
    const float* __restrict__ bv,
    const unsigned int* __restrict__ Wbf,   // 4 matrices, 512K dwords each
    unsigned short* __restrict__ Qbf, unsigned short* __restrict__ Kbf,
    unsigned short* __restrict__ Vt) {
  __shared__ unsigned int WB[2][64 * WB_STRIDE];   // 2 x 5 KB weight panels

  const int tid  = threadIdx.x;
  const int lane = tid & 31;
  const int w    = tid >> 5;                   // wave in block (0..7)
  const int bid  = blockIdx.x;                 // 0..1535
  const int proj = bid >> 9;                   // 512 blocks per projection
  const int rem  = bid & 511;
  const int nt   = rem >> 5;                   // col block (64 cols)
  const int mtg  = rem & 31;                   // row-tile group (8 tiles)
  const int mt   = mtg * 8 + w;                // this wave's 16-row tile
  const int m    = lane & 15;
  const int hf   = lane >> 4;
  const int n    = m;

  const float* bias = (proj == 0) ? bq : (proj == 1) ? bk : bv;
  const unsigned int* W    = Wbf + (size_t)proj * (1u << 19);
  const unsigned int* Arow = Xbf + (size_t)proj * (1u << 21)
                                 + (size_t)(mt * 16 + m) * (DD / 2);

  // staging role of this thread: one b128 (4 dwords) per k-step
  const int srow = tid >> 2;                   // 0..63  (weight row in panel)
  const int sseg = tid & 3;                    // 0..3   (16B segment)
  const unsigned int* Wsrc0 =
      W + (size_t)(nt * 64 + srow) * (DD / 2) + sseg * 4;

  v8f acc[4];
#pragma unroll
  for (int i = 0; i < 4; ++i) acc[i] = v8f_zero();

  // prologue: stage k-step 0 into buffer 0
  copy_b128_to_lds(Wsrc0, &WB[0][srow * WB_STRIDE + sseg * 4]);

  for (int kt = 0; kt < 32; ++kt) {
    const int buf = kt & 1;
    __syncthreads();                 // readers of buf^1 (iter kt-1) are done
    if (kt + 1 < 32)
      copy_b128_to_lds(Wsrc0 + (kt + 1) * 16,
                       &WB[buf ^ 1][srow * WB_STRIDE + sseg * 4]);
    if (kt + 1 < 32) { WAIT_ASYNC(1); } else { WAIT_ASYNC(0); }
    __syncthreads();                 // everyone's buf[kt] panel is visible

    // A tile: 16x32 bf16; dword = kt*16 + hf*8 + v
    v8u au;
#pragma unroll
    for (int v = 0; v < 8; ++v)
      au[v] = Arow[kt * 16 + hf * 8 + v];
    const v16bf a = as_bf(au);

#pragma unroll
    for (int nb = 0; nb < 4; ++nb) {
      // B tile from LDS: row = nb*16+n, k0 = hf*16+2v -> dword hf*8+v
      const unsigned int* Brow = &WB[buf][(nb * 16 + n) * WB_STRIDE + hf * 8];
      v8u bu;
#pragma unroll
      for (int v = 0; v < 8; ++v) bu[v] = Brow[v];
      acc[nb] = WMMA_BF16(a, as_bf(bu), acc[nb]);
    }
  }

  // C layout: VGPR r -> row (r + 8*hf), col n.  Add bias, store.
#pragma unroll
  for (int nb = 0; nb < 4; ++nb) {
    const int o   = nt * 64 + nb * 16 + n;
    const float bv_ = bias[o];
    const int h = o >> 6, dh = o & 63;
#pragma unroll
    for (int r = 0; r < 8; ++r) {
      const int i = mt * 16 + r + 8 * hf;          // global row in [B*L]
      const int b = i >> 11, l = i & (LL - 1);
      const float val = acc[nb][r] + bv_;
      if (proj == 2) {
        Vt[((size_t)((b * HH + h) * DK + dh)) * LL + l] = cvt1_bf16(val);
      } else {
        unsigned short* O = (proj == 0) ? Qbf : Kbf;
        O[((size_t)((b * HH + h) * LL + l)) * DK + dh] = cvt1_bf16(val);
      }
    }
  }
}

// ---------------------------------------------------------------------------
// Kernel 2: attention with online (flash) softmax.
// One wave per (b,h, 16-query tile).  Key loop in blocks of 256; 16 KB LDS
// score slab; additive 0/1 band mask (reference ADDS the float mask).
// ---------------------------------------------------------------------------
__global__ __launch_bounds__(32) void attn_kernel(
    const unsigned short* __restrict__ Qbf, const unsigned short* __restrict__ Kbf,
    const unsigned short* __restrict__ Vt, unsigned short* __restrict__ Obf,
    const int* __restrict__ masksize_p) {
  __shared__ float Sblk[16 * 256];                 // 16 KB

  const int lane = threadIdx.x & 31;
  const int m  = lane & 15;
  const int hf = lane >> 4;
  const int wg = blockIdx.x;                       // 0..4095
  const int qt = wg & 127;                         // query tile
  const int bh = wg >> 7;                          // b*H + h
  const int half = masksize_p[0] >> 1;
  const float scale = 0.125f;                      // 1/sqrt(64)

  // Q A-tiles resident in registers: dk=64 -> two 16x32 chunks.
  const unsigned int* Qrow = reinterpret_cast<const unsigned int*>(
      Qbf + (size_t)bh * LL * DK + (size_t)(qt * 16 + m) * DK);
  v16bf qa[2];
#pragma unroll
  for (int c = 0; c < 2; ++c) {
    v8u u;
#pragma unroll
    for (int v = 0; v < 8; ++v)
      u[v] = Qrow[c * 16 + hf * 8 + v];
    qa[c] = as_bf(u);
  }

  v8f o_acc[4];
#pragma unroll
  for (int i = 0; i < 4; ++i) o_acc[i] = v8f_zero();
  float run_max = -__builtin_inff();
  float run_sum = 0.0f;

  const unsigned int* Kbase =
      reinterpret_cast<const unsigned int*>(Kbf + (size_t)bh * LL * DK);
  const unsigned int* Vbase =
      reinterpret_cast<const unsigned int*>(Vt + (size_t)bh * DK * LL);

  for (int blk = 0; blk < 8; ++blk) {
    // ---- scores for 256 keys (16 WMMA tiles, 2 k-chunks each) -------------
    for (int kt = 0; kt < 16; ++kt) {
      const int key = blk * 256 + kt * 16 + m;     // B col = lane&15
      const unsigned int* Krow = Kbase + (size_t)key * (DK / 2);
      v8f s = v8f_zero();
#pragma unroll
      for (int c = 0; c < 2; ++c) {
        v8u u;
#pragma unroll
        for (int v = 0; v < 8; ++v) u[v] = Krow[c * 16 + hf * 8 + v];
        s = WMMA_BF16(qa[c], as_bf(u), s);
      }
#pragma unroll
      for (int r = 0; r < 8; ++r) {
        const int q = qt * 16 + r + 8 * hf;
        const int k = blk * 256 + kt * 16 + m;
        int d = q - k; d = (d < 0) ? -d : d;
        const float band = (d <= half) ? 1.0f : 0.0f;
        Sblk[(r + 8 * hf) * 256 + kt * 16 + m] = s[r] * scale + band;
      }
    }
    __syncthreads();

    // ---- online softmax over this block (lane: row m, half hf) ------------
    float* Srow = &Sblk[m * 256 + hf * 128];
    float bm = -__builtin_inff();
#pragma unroll 4
    for (int j = 0; j < 128; ++j) bm = fmaxf(bm, Srow[j]);
    bm = fmaxf(bm, __shfl_xor(bm, 16, 32));
    const float nm = fmaxf(run_max, bm);
    const float corr = __expf(run_max - nm);
    float bsum = 0.0f;
#pragma unroll 4
    for (int j = 0; j < 128; ++j) {
      const float e = __expf(Srow[j] - nm);
      Srow[j] = e;
      bsum += e;
    }
    bsum += __shfl_xor(bsum, 16, 32);
    run_sum = run_sum * corr + bsum;
    run_max = nm;
    __syncthreads();

    // ---- rescale accumulators by per-row corr (broadcast from lanes 0..15)
#pragma unroll
    for (int r = 0; r < 8; ++r) {
      const float cr = __shfl(corr, r + 8 * hf, 32);
#pragma unroll
      for (int nb = 0; nb < 4; ++nb) o_acc[nb][r] *= cr;
    }

    // ---- P @ V over this block: 8 chunks of 32 keys ------------------------
    for (int c2 = 0; c2 < 8; ++c2) {
      v8u pu;
#pragma unroll
      for (int v = 0; v < 8; ++v) {
        const int k0 = c2 * 32 + hf * 16 + v * 2;
        float2 p2 = *reinterpret_cast<const float2*>(&Sblk[m * 256 + k0]);
        pu[v] = pack2_bf16(p2.x, p2.y);
      }
      const v16bf pa = as_bf(pu);
#pragma unroll
      for (int nb = 0; nb < 4; ++nb) {
        const int feat = nb * 16 + m;              // B col = feature
        const unsigned int* Vrow = Vbase + (size_t)feat * (LL / 2);
        v8u vu;
#pragma unroll
        for (int v = 0; v < 8; ++v)
          vu[v] = Vrow[blk * 128 + c2 * 16 + hf * 8 + v];
        o_acc[nb] = WMMA_BF16(pa, as_bf(vu), o_acc[nb]);
      }
    }
    __syncthreads();   // Sblk reused next block
  }

  // ---- normalize by 1/sum and store as [B*L, D] bf16 ------------------------
  const float inv = 1.0f / run_sum;                // valid for row m
  const int b = bh >> 4, h = bh & 15;
#pragma unroll
  for (int r = 0; r < 8; ++r) {
    const float ir = __shfl(inv, r + 8 * hf, 32);
    const int l = qt * 16 + r + 8 * hf;
#pragma unroll
    for (int nb = 0; nb < 4; ++nb) {
      const int feat = nb * 16 + m;
      Obf[(size_t)(b * LL + l) * DD + h * DK + feat] =
          cvt1_bf16(o_acc[nb][r] * ir);
    }
  }
}

// ---------------------------------------------------------------------------
// Kernel 3: output projection  Y = Obf(bf16) @ Wo^T(bf16) + bo  -> f32 d_out.
// ---------------------------------------------------------------------------
__global__ __launch_bounds__(256) void out_proj_kernel(
    const unsigned short* __restrict__ Obf, const unsigned int* __restrict__ Wbf,
    const float* __restrict__ bo, float* __restrict__ Y) {
  const int lane = threadIdx.x & 31;
  const int wave = blockIdx.x * (blockDim.x >> 5) + (threadIdx.x >> 5); // 0..4095
  const int mt = wave & 255;
  const int nt = wave >> 8;
  const int m  = lane & 15;
  const int hf = lane >> 4;
  const int n  = m;

  const unsigned int* Wo = Wbf + (size_t)3 * (1u << 19);   // slot 3
  const unsigned int* Arow =
      reinterpret_cast<const unsigned int*>(Obf + (size_t)(mt * 16 + m) * DD);

  v8f acc[4];
#pragma unroll
  for (int i = 0; i < 4; ++i) acc[i] = v8f_zero();

  for (int kt = 0; kt < 32; ++kt) {
    v8u au;
#pragma unroll
    for (int v = 0; v < 8; ++v)
      au[v] = Arow[kt * 16 + hf * 8 + v];
    const v16bf a = as_bf(au);
#pragma unroll
    for (int nb = 0; nb < 4; ++nb) {
      const int o = nt * 64 + nb * 16 + n;
      const unsigned int* Wrow = Wo + (size_t)o * 512 + kt * 16 + hf * 8;
      v8u bu;
#pragma unroll
      for (int v = 0; v < 8; ++v) bu[v] = Wrow[v];
      acc[nb] = WMMA_BF16(a, as_bf(bu), acc[nb]);
    }
  }

#pragma unroll
  for (int nb = 0; nb < 4; ++nb) {
    const int o = nt * 64 + nb * 16 + n;
    const float bo_ = bo[o];
#pragma unroll
    for (int r = 0; r < 8; ++r) {
      const int i = mt * 16 + r + 8 * hf;
      Y[(size_t)i * DD + o] = acc[nb][r] + bo_;
    }
  }
}

// ---------------------------------------------------------------------------
// Launch
// ---------------------------------------------------------------------------
extern "C" void kernel_launch(void* const* d_in, const int* in_sizes, int n_in,
                              void* d_out, int out_size, void* d_ws, size_t ws_size,
                              hipStream_t stream) {
  (void)in_sizes; (void)n_in; (void)out_size; (void)ws_size;
  const float* q_in = (const float*)d_in[0];
  const float* k_in = (const float*)d_in[1];
  const float* v_in = (const float*)d_in[2];
  const float* Wq   = (const float*)d_in[3];
  const float* bq   = (const float*)d_in[4];
  const float* Wk   = (const float*)d_in[5];
  const float* bk   = (const float*)d_in[6];
  const float* Wv   = (const float*)d_in[7];
  const float* bv   = (const float*)d_in[8];
  const float* Wo   = (const float*)d_in[9];
  const float* bo   = (const float*)d_in[10];
  const int*   msk  = (const int*)d_in[11];

  char* ws = (char*)d_ws;
  unsigned short* Qbf = (unsigned short*)(ws);                       // 8 MiB
  unsigned short* Kbf = (unsigned short*)(ws + ((size_t)8  << 20));  // 8 MiB
  unsigned short* Vt  = (unsigned short*)(ws + ((size_t)16 << 20));  // 8 MiB
  unsigned short* Obf = (unsigned short*)(ws + ((size_t)24 << 20));  // 8 MiB
  unsigned int*   Wbf = (unsigned int*)  (ws + ((size_t)32 << 20));  // 8 MiB
  unsigned int*   Xbf = (unsigned int*)  (ws + ((size_t)40 << 20));  // 24 MiB

  // 0) weights + activations f32 -> bf16
  cvt_w_kernel<<<8192, 256, 0, stream>>>(Wq, Wk, Wv, Wo, Wbf);
  cvt_x_kernel<<<24576, 256, 0, stream>>>(q_in, k_in, v_in, Xbf);
  // 1) Q/K/V projections: 3 x (16 col panels x 32 row groups) blocks
  qkv_proj_kernel<<<1536, 256, 0, stream>>>(Xbf, bq, bk, bv, Wbf, Qbf, Kbf, Vt);
  // 2) attention: one wave per (b,h,16-query tile)
  attn_kernel<<<BB * HH * (LL / 16), 32, 0, stream>>>(Qbf, Kbf, Vt, Obf, msk);
  // 3) output projection: 4096 wave tiles, 8 waves/block
  out_proj_kernel<<<512, 256, 0, stream>>>(Obf, Wbf, bo, (float*)d_out);
}
